// LiquidNNSensorModel_27848567947564
// MI455X (gfx1250) — matrix operand order repaired and checked by reading, not compiled, not run on hardware
//
#include <hip/hip_runtime.h>
#include <hip/hip_bf16.h>
#include <math.h>

// ---------------- problem constants ----------------
#define BB   256   // batch
#define TT   1024  // timesteps
#define SS   8     // sensors
#define HH   256   // hidden
#define LL   32    // latent
#define RHH  32    // risk hidden

typedef __attribute__((ext_vector_type(16))) __bf16        v16bf;
typedef __attribute__((ext_vector_type(8)))  float         v8f;
typedef __attribute__((ext_vector_type(4)))  unsigned int  u32x4;

#define GLOBAL_AS __attribute__((address_space(1)))

// packed-weight element offsets (in bf16 elements)
#define OFF_GX0   0        // 16 tiles (K=32 padded from S=8)
#define OFF_BBX0  8192     // 16 tiles
#define OFF_GH0   16384    // 128 tiles (K=256)
#define OFF_BBH0  81920
#define OFF_GX1   147456
#define OFF_BB1A  212992
#define OFF_GH1   278528
#define OFF_BB1B  344064
#define WPK_TOTAL 409600

// workspace byte offsets
#define WS_XLN   0u                       // [T][B][S] bf16  = 4,194,304 B
#define WS_DT    4194304u                 // [T][B] f32      = 1,048,576 B
#define WS_WPK   5242880u                 // packed bf16     =   819,200 B
#define WS_BIAS  6062080u                 // 6*256 f32       =     6,144 B
#define WS_H1    6068224u                 // [B][H] f32      =   262,144 B
// total ~6.33 MB

// ------------------------------------------------------------------
// Prep 1: LayerNorm inputs -> bf16 [t][b][s], dt -> f32 [t][b]
// ------------------------------------------------------------------
__global__ __launch_bounds__(256) void prep_x_dt(
    const float* __restrict__ x, const float* __restrict__ ts,
    const float* __restrict__ ln_g, const float* __restrict__ ln_b,
    __bf16* __restrict__ xln, float* __restrict__ dtg)
{
  int id = blockIdx.x * 256 + threadIdx.x;
  if (id >= BB * TT) return;
  int b = id / TT, t = id - b * TT;
  const float* xr = x + ((size_t)b * TT + t) * SS;
  float mu = 0.f;
#pragma unroll
  for (int s = 0; s < SS; ++s) mu += xr[s];
  mu *= (1.f / SS);
  float var = 0.f;
#pragma unroll
  for (int s = 0; s < SS; ++s) { float d = xr[s] - mu; var += d * d; }
  var *= (1.f / SS);
  float r = rsqrtf(var + 1e-5f);
  __bf16* o = xln + ((size_t)t * BB + b) * SS;
#pragma unroll
  for (int s = 0; s < SS; ++s)
    o[s] = (__bf16)((xr[s] - mu) * r * ln_g[s] + ln_b[s]);
  float d = (t == 0) ? 1.f : fmaxf(ts[(size_t)b * TT + t] - ts[(size_t)b * TT + t - 1], 1e-6f);
  dtg[(size_t)t * BB + b] = d;
}

// ------------------------------------------------------------------
// Prep 2: pack fp32 weights into bf16 WMMA B-tile layout.
// B-tile (32K x 16N), per-lane order: lane = n%16 + 16*((k/8)%2 within tile),
// element idx = k%8 + 8*(k/16).  Tile storage: base + (nt*KT + kt)*512, lane*16 + idx.
// ------------------------------------------------------------------
__global__ __launch_bounds__(256) void prep_pack(
    const float* __restrict__ gx_w0, const float* __restrict__ bb_w0,
    const float* __restrict__ gh_w0, const float* __restrict__ gx_w1,
    const float* __restrict__ gh_w1, const float* __restrict__ bb_w1,
    __bf16* __restrict__ wpk)
{
  int e = blockIdx.x * 256 + threadIdx.x;
  if (e >= WPK_TOTAL) return;
  const float* W; int ldw, kofs, Kvalid, KTm, rel;
  if      (e < 16384)  { if (e < 8192) { W = gx_w0; ldw = 8;   kofs = 0;   rel = e; }
                         else          { W = bb_w0; ldw = 264; kofs = 0;   rel = e - 8192; }
                         Kvalid = 8; KTm = 1; }
  else if (e < 81920)  { W = gh_w0; ldw = 256; kofs = 0;   Kvalid = 256; KTm = 8; rel = e - 16384; }
  else if (e < 147456) { W = bb_w0; ldw = 264; kofs = 8;   Kvalid = 256; KTm = 8; rel = e - 81920; }
  else if (e < 212992) { W = gx_w1; ldw = 256; kofs = 0;   Kvalid = 256; KTm = 8; rel = e - 147456; }
  else if (e < 278528) { W = bb_w1; ldw = 512; kofs = 0;   Kvalid = 256; KTm = 8; rel = e - 212992; }
  else if (e < 344064) { W = gh_w1; ldw = 256; kofs = 0;   Kvalid = 256; KTm = 8; rel = e - 278528; }
  else                 { W = bb_w1; ldw = 512; kofs = 256; Kvalid = 256; KTm = 8; rel = e - 344064; }
  int tile = rel >> 9, within = rel & 511;
  int lane = within >> 4, idx = within & 15;
  int nt = tile / KTm, kt = tile - nt * KTm;
  int n = nt * 16 + (lane & 15);
  int k = kt * 32 + ((lane >> 4) << 3) + (idx & 7) + ((idx >> 3) << 4);
  float v = (k < Kvalid) ? W[n * ldw + kofs + k] : 0.f;
  wpk[e] = (__bf16)v;
}

// ------------------------------------------------------------------
// Prep 3: fused biases + softplus(log-tau)
// bias layout: [bg0 | bf0 | bg1 | bf1 | sp0 | sp1], each 256 f32
// ------------------------------------------------------------------
__global__ __launch_bounds__(256) void prep_bias(
    const float* gx_b0, const float* gb0, const float* bb_b0, const float* lt0,
    const float* gx_b1, const float* gb1, const float* bb_b1, const float* lt1,
    float* __restrict__ bias)
{
  int n = threadIdx.x;
  bias[n]        = gx_b0[n] + gb0[n];
  bias[256 + n]  = bb_b0[n];
  bias[512 + n]  = gx_b1[n] + gb1[n];
  bias[768 + n]  = bb_b1[n];
  float l0 = lt0[n]; bias[1024 + n] = (l0 > 20.f) ? l0 : log1pf(expf(l0));
  float l1 = lt1[n]; bias[1280 + n] = (l1 > 20.f) ? l1 : log1pf(expf(l1));
}

// ------------------------------------------------------------------
// WMMA helpers
// ------------------------------------------------------------------
__device__ __forceinline__ v8f wmma_bf16(v16bf a, v16bf b, v8f c) {
  return __builtin_amdgcn_wmma_f32_16x16x32_bf16(
      /*neg_a=*/false, a, /*neg_b=*/false, b,
      /*c_mod=*/(short)0, c, /*reuse_a=*/false, /*reuse_b=*/false);
}

// B operand: 32 contiguous bytes per lane from a packed tile in GLOBAL memory.
// AS(1) pointer => global_load_b128 saddr+imm form (LOADcnt only, no DScnt).
__device__ __forceinline__ v16bf load_b_tile(const GLOBAL_AS __bf16* p, int lane) {
  union { u32x4 q[2]; v16bf v; } u;
  const GLOBAL_AS u32x4* lp = (const GLOBAL_AS u32x4*)(p + lane * 16);
  u.q[0] = lp[0];
  u.q[1] = lp[1];
  return u.v;
}

// A operand (16-bit A 16x32, ISA 7.12.2): lane m<16 holds K {0..7,16..23},
// lane m+16 holds K {8..15,24..31}; rowp = &h[m][kt*32 + hi*8]
__device__ __forceinline__ v16bf load_a_tile(const __bf16* rowp) {
  union { u32x4 q[2]; v16bf v; } u;
  u.q[0] = *(const u32x4*)(rowp);
  u.q[1] = *(const u32x4*)(rowp + 16);
  return u.v;
}

// fast transcendentals (single v_exp_f32-based, saturating-correct)
__device__ __forceinline__ float fexp(float x)  { return __expf(x); }
__device__ __forceinline__ float fsigm(float x) { return 1.f / (1.f + __expf(-x)); }
__device__ __forceinline__ float ftanh(float x) { return 1.f - 2.f / (__expf(2.f * x) + 1.f); }

// CfC elementwise update for one 16x16 output tile pair (g,f); C-layout:
// vgpr j -> row m = j + 8*hi, lane -> col n.  Writes bf16 h to LDS.
__device__ __forceinline__ v8f cfc_update(v8f cg, v8f cf, v8f hold, v8f dtv,
                                          float bg, float bf, float sp,
                                          __bf16 (*hA)[HH], int n, int hi) {
  v8f hnew;
#pragma unroll
  for (int j = 0; j < 8; ++j) {
    float g   = fsigm(cg[j] + bg);
    float f   = ftanh(cf[j] + bf);
    float tau = sp + fabsf(g);
    float dec = fexp(-dtv[j] * tau);
    float hn  = dec * hold[j] + (1.f - dec) * f;
    hnew[j] = hn;
    hA[j + 8 * hi][n] = (__bf16)hn;
  }
  return hnew;
}

// ------------------------------------------------------------------
// Recurrent CfC kernel: 16 blocks x 256 threads (8 waves).
// Block owns 16 batch rows; wave owns 2 N-tiles (32 hidden cols) of both
// layers; persistent h slices live in VGPRs in WMMA C-layout.
// Weight tiles are re-streamed from L2 every step: the packed-weight base
// address (a u64, re-materialized as an addrspace(1) pointer) is opacified
// per iteration so LICM cannot hoist ~100 B-tiles into registers, while the
// loads still lower to global_load_b128 saddr+imm (not flat).
// ------------------------------------------------------------------
__global__ __launch_bounds__(256) void cfc_recurrent(
    const __bf16* __restrict__ xln, const float* __restrict__ dtg,
    const __bf16* __restrict__ wpk, const float* __restrict__ bias,
    float* __restrict__ h1out)
{
  __shared__ __align__(32) __bf16 hA0[16][HH];  // h0 (bf16, row-major)
  __shared__ __align__(32) __bf16 hA1[16][HH];  // h1

  const int tid  = threadIdx.x;
  const int lane = tid & 31;
  const int wave = tid >> 5;
  const int hi   = lane >> 4;
  const int ln   = lane & 15;
  const int nt0  = wave * 2, nt1 = wave * 2 + 1;
  const int n0   = nt0 * 16 + ln, n1 = nt1 * 16 + ln;
  const int wgb  = blockIdx.x * 16;   // batch row base
  const int mrow = ln;                // A-operand row for this lane

  // per-lane fused-bias / softplus constants
  const float bg0_0 = bias[n0],        bg0_1 = bias[n1];
  const float bf0_0 = bias[256 + n0],  bf0_1 = bias[256 + n1];
  const float bg1_0 = bias[512 + n0],  bg1_1 = bias[512 + n1];
  const float bf1_0 = bias[768 + n0],  bf1_1 = bias[768 + n1];
  const float sp0_0 = bias[1024 + n0], sp0_1 = bias[1024 + n1];
  const float sp1_0 = bias[1280 + n0], sp1_1 = bias[1280 + n1];

  // per-wave weight stream element offsets (loop-invariant ints, cheap)
  const int oGx0a  = OFF_GX0  + nt0 * 512;
  const int oGx0b  = OFF_GX0  + nt1 * 512;
  const int oBbx0a = OFF_BBX0 + nt0 * 512;
  const int oBbx0b = OFF_BBX0 + nt1 * 512;
  const int oGh0a  = OFF_GH0  + nt0 * 8 * 512;
  const int oGh0b  = OFF_GH0  + nt1 * 8 * 512;
  const int oBbh0a = OFF_BBH0 + nt0 * 8 * 512;
  const int oBbh0b = OFF_BBH0 + nt1 * 8 * 512;
  const int oGx1a  = OFF_GX1  + nt0 * 8 * 512;
  const int oGx1b  = OFF_GX1  + nt1 * 8 * 512;
  const int oBb1aa = OFF_BB1A + nt0 * 8 * 512;
  const int oBb1ab = OFF_BB1A + nt1 * 8 * 512;
  const int oGh1a  = OFF_GH1  + nt0 * 8 * 512;
  const int oGh1b  = OFF_GH1  + nt1 * 8 * 512;
  const int oBb1ba = OFF_BB1B + nt0 * 8 * 512;
  const int oBb1bb = OFF_BB1B + nt1 * 8 * 512;

  // persistent own h slices (f32, C-layout): h0 tiles nt0/nt1, h1 tiles nt0/nt1
  v8f h0a = {}, h0b = {}, h1a = {}, h1b = {};

  // zero initial h in LDS
  for (int i = tid; i < 16 * HH; i += 256) {
    (&hA0[0][0])[i] = (__bf16)0.f;
    (&hA1[0][0])[i] = (__bf16)0.f;
  }
  __syncthreads();

  unsigned long long wbase = (unsigned long long)wpk;
  for (int t = 0; t < TT; ++t) {
    // Opacify the weight base: forces B-tiles to be re-streamed from L2
    // each step instead of hoisted + spilled; integer base keeps the
    // addrspace(1) cast below intact => global_load (LOADcnt-only path).
    asm volatile("" : "+s"(wbase));
    const GLOBAL_AS __bf16* wq = (const GLOBAL_AS __bf16*)wbase;

    // dt for this half's 8 batch rows (dtv[j] = dt[wgb + j + 8*hi])
    v8f dtv = *(const v8f*)(dtg + (size_t)t * BB + wgb + hi * 8);

    // ------------- layer 0 -------------
    v8f cg0 = {}, cg1 = {}, cf0 = {}, cf1 = {};
    {
      // x contribution: K=32 tile, rows 0..7 = layer-normed sensors, rest 0
      union { u32x4 q[2]; v16bf v; } ua;
      ua.q[0] = (u32x4)0u;
      ua.q[1] = (u32x4)0u;
      if (hi == 0)   // lo lanes: elements 0..7 = K0..7 (sensors); K16..23 = 0
        ua.q[0] = *(const u32x4*)(xln + ((size_t)t * BB + wgb + ln) * SS);
      v16bf ax = ua.v;
      cg0 = wmma_bf16(ax, load_b_tile(wq + oGx0a,  lane), cg0);
      cg1 = wmma_bf16(ax, load_b_tile(wq + oGx0b,  lane), cg1);
      cf0 = wmma_bf16(ax, load_b_tile(wq + oBbx0a, lane), cf0);
      cf1 = wmma_bf16(ax, load_b_tile(wq + oBbx0b, lane), cf1);
    }
#pragma unroll
    for (int kt = 0; kt < 8; ++kt) {
      v16bf a = load_a_tile(&hA0[mrow][kt * 32 + hi * 8]);
      cg0 = wmma_bf16(a, load_b_tile(wq + oGh0a  + kt * 512, lane), cg0);
      cg1 = wmma_bf16(a, load_b_tile(wq + oGh0b  + kt * 512, lane), cg1);
      cf0 = wmma_bf16(a, load_b_tile(wq + oBbh0a + kt * 512, lane), cf0);
      cf1 = wmma_bf16(a, load_b_tile(wq + oBbh0b + kt * 512, lane), cf1);
    }
    __syncthreads();   // all reads of hA0(old) complete
    h0a = cfc_update(cg0, cf0, h0a, dtv, bg0_0, bf0_0, sp0_0, hA0, n0, hi);
    h0b = cfc_update(cg1, cf1, h0b, dtv, bg0_1, bf0_1, sp0_1, hA0, n1, hi);
    __syncthreads();   // hA0 now holds h0(new)

    // ------------- layer 1 -------------
    cg0 = (v8f){}; cg1 = (v8f){}; cf0 = (v8f){}; cf1 = (v8f){};
#pragma unroll
    for (int kt = 0; kt < 8; ++kt) {   // A = h0(new)
      v16bf a = load_a_tile(&hA0[mrow][kt * 32 + hi * 8]);
      cg0 = wmma_bf16(a, load_b_tile(wq + oGx1a  + kt * 512, lane), cg0);
      cg1 = wmma_bf16(a, load_b_tile(wq + oGx1b  + kt * 512, lane), cg1);
      cf0 = wmma_bf16(a, load_b_tile(wq + oBb1aa + kt * 512, lane), cf0);
      cf1 = wmma_bf16(a, load_b_tile(wq + oBb1ab + kt * 512, lane), cf1);
    }
#pragma unroll
    for (int kt = 0; kt < 8; ++kt) {   // A = h1(old)
      v16bf a = load_a_tile(&hA1[mrow][kt * 32 + hi * 8]);
      cg0 = wmma_bf16(a, load_b_tile(wq + oGh1a  + kt * 512, lane), cg0);
      cg1 = wmma_bf16(a, load_b_tile(wq + oGh1b  + kt * 512, lane), cg1);
      cf0 = wmma_bf16(a, load_b_tile(wq + oBb1ba + kt * 512, lane), cf0);
      cf1 = wmma_bf16(a, load_b_tile(wq + oBb1bb + kt * 512, lane), cf1);
    }
    __syncthreads();   // all reads of hA1(old) complete
    h1a = cfc_update(cg0, cf0, h1a, dtv, bg1_0, bf1_0, sp1_0, hA1, n0, hi);
    h1b = cfc_update(cg1, cf1, h1b, dtv, bg1_1, bf1_1, sp1_1, hA1, n1, hi);
    // no barrier needed here: the next step's two phase-0 barriers already
    // separate these hA1 writes from the next hA1 reads (phase 1 of t+1).
  }

  // write final h1 (f32) for the head
#pragma unroll
  for (int j = 0; j < 8; ++j) {
    int m = wgb + j + 8 * hi;
    h1out[(size_t)m * HH + n0] = h1a[j];
    h1out[(size_t)m * HH + n1] = h1b[j];
  }
}

// ------------------------------------------------------------------
// Head: latent = tanh(h1 @ lp_w.T + lp_b); gelu(exact erf); sigmoid
// out = [latent (B*L) | risk (B)]
// ------------------------------------------------------------------
__global__ __launch_bounds__(256) void head_kernel(
    const float* __restrict__ h1, const float* __restrict__ lp_w,
    const float* __restrict__ lp_b, const float* __restrict__ r1_w,
    const float* __restrict__ r1_b, const float* __restrict__ r2_w,
    const float* __restrict__ r2_b, float* __restrict__ out)
{
  int b = blockIdx.x * 256 + threadIdx.x;
  if (b >= BB) return;
  const float* hr = h1 + (size_t)b * HH;
  float lat[LL];
#pragma unroll 4
  for (int l = 0; l < LL; ++l) {
    float acc = lp_b[l];
    for (int k = 0; k < HH; ++k) acc += lp_w[l * HH + k] * hr[k];
    lat[l] = tanhf(acc);
    out[(size_t)b * LL + l] = lat[l];
  }
  float hid[RHH];
#pragma unroll 4
  for (int j = 0; j < RHH; ++j) {
    float acc = r1_b[j];
    for (int k = 0; k < LL; ++k) acc += r1_w[j * LL + k] * lat[k];
    hid[j] = 0.5f * acc * (1.f + erff(acc * 0.70710678118654752f));  // exact GELU
  }
  float acc = r2_b[0];
  for (int k = 0; k < RHH; ++k) acc += r2_w[k] * hid[k];
  out[BB * LL + b] = 1.f / (1.f + expf(-acc));
}

// ------------------------------------------------------------------
extern "C" void kernel_launch(void* const* d_in, const int* in_sizes, int n_in,
                              void* d_out, int out_size, void* d_ws, size_t ws_size,
                              hipStream_t stream) {
  const float* sensor = (const float*)d_in[0];
  const float* tstamp = (const float*)d_in[1];
  const float* ln_g   = (const float*)d_in[2];
  const float* ln_b   = (const float*)d_in[3];
  const float* bb_w0  = (const float*)d_in[4];
  const float* bb_b0  = (const float*)d_in[5];
  const float* gx_w0  = (const float*)d_in[6];
  const float* gx_b0  = (const float*)d_in[7];
  const float* gh_w0  = (const float*)d_in[8];
  const float* gb0    = (const float*)d_in[9];
  const float* lt0    = (const float*)d_in[10];
  const float* bb_w1  = (const float*)d_in[11];
  const float* bb_b1  = (const float*)d_in[12];
  const float* gx_w1  = (const float*)d_in[13];
  const float* gx_b1  = (const float*)d_in[14];
  const float* gh_w1  = (const float*)d_in[15];
  const float* gb1    = (const float*)d_in[16];
  const float* lt1    = (const float*)d_in[17];
  const float* lp_w   = (const float*)d_in[18];
  const float* lp_b   = (const float*)d_in[19];
  const float* r1_w   = (const float*)d_in[20];
  const float* r1_b   = (const float*)d_in[21];
  const float* r2_w   = (const float*)d_in[22];
  const float* r2_b   = (const float*)d_in[23];

  char* ws = (char*)d_ws;
  __bf16* xln  = (__bf16*)(ws + WS_XLN);
  float*  dtg  = (float*)(ws + WS_DT);
  __bf16* wpk  = (__bf16*)(ws + WS_WPK);
  float*  bias = (float*)(ws + WS_BIAS);
  float*  h1f  = (float*)(ws + WS_H1);
  float*  out  = (float*)d_out;

  prep_x_dt<<<(BB * TT + 255) / 256, 256, 0, stream>>>(sensor, tstamp, ln_g, ln_b, xln, dtg);
  prep_pack<<<(WPK_TOTAL + 255) / 256, 256, 0, stream>>>(gx_w0, bb_w0, gh_w0, gx_w1, gh_w1, bb_w1, wpk);
  prep_bias<<<1, 256, 0, stream>>>(gx_b0, gb0, bb_b0, lt0, gx_b1, gb1, bb_b1, lt1, bias);
  cfc_recurrent<<<BB / 16, 256, 0, stream>>>(xln, dtg, wpk, bias, h1f);
  head_kernel<<<1, 256, 0, stream>>>(h1f, lp_w, lp_b, r1_w, r1_b, r2_w, r2_b, out);
}
